// UPTRec_10024453669716
// MI455X (gfx1250) — matrix operand order repaired and checked by reading, not compiled, not run on hardware
//
#include <hip/hip_runtime.h>
#include <stdint.h>

#define B_ 512
#define T_ 256
#define C_ 256
#define H_ 4
#define DH_ 64
#define L_ 2
#define KC_ 16
#define KM_ITERS 10
#define ITEM_H_ 192
#define USER_H_ 64
#define BT_ (B_*T_)
static const size_t N_ = (size_t)BT_ * C_;

typedef __attribute__((ext_vector_type(16))) __bf16 v16bf;
typedef __attribute__((ext_vector_type(8)))  float  v8f;
typedef unsigned int  u32;
typedef unsigned short u16;

__device__ __forceinline__ u16 f2bf(float x) {
    u32 u = __float_as_uint(x);
    u32 r = (u + 0x7FFFu + ((u >> 16) & 1u)) >> 16;
    return (u16)r;
}
__device__ __forceinline__ float bf2f(u16 h) {
    return __uint_as_float(((u32)h) << 16);
}
__device__ __forceinline__ u32 pack2(float lo, float hi) {
    return (u32)f2bf(lo) | ((u32)f2bf(hi) << 16);
}
// reductions within each 16-lane half-wave (stay inside the C/D-layout row group)
__device__ __forceinline__ float rmax16(float v) {
#pragma unroll
    for (int o = 1; o < 16; o <<= 1) v = fmaxf(v, __shfl_xor(v, o, 32));
    return v;
}
__device__ __forceinline__ float rsum16(float v) {
#pragma unroll
    for (int o = 1; o < 16; o <<= 1) v += __shfl_xor(v, o, 32);
    return v;
}

// ---------------------------------------------------------------- embed
__global__ void embed_kernel(const int* __restrict__ uid, const int* __restrict__ seq,
                             const float* __restrict__ item_emb, const float* __restrict__ user_emb,
                             const float* __restrict__ pos_emb, float* __restrict__ X,
                             int* __restrict__ pad) {
    size_t i = (size_t)blockIdx.x * blockDim.x + threadIdx.x;
    if (i >= N_) return;
    int row = (int)(i / C_);
    int j   = (int)(i % C_);
    int b = row / T_;
    int t = row % T_;
    int s = seq[row];
    float v;
    if (j < ITEM_H_) v = item_emb[(size_t)s * ITEM_H_ + j];
    else             v = user_emb[(size_t)uid[b] * USER_H_ + (j - ITEM_H_)];
    v += pos_emb[t * C_ + j];
    X[i] = v;
    if (j == 0) pad[row] = (s == 0);
}

__global__ void ulat_kernel(const int* __restrict__ uid, const float* __restrict__ user_emb,
                            float* __restrict__ ulat) {
    int i = blockIdx.x * blockDim.x + threadIdx.x;
    if (i >= B_ * USER_H_) return;
    int b = i / USER_H_;
    int j = i % USER_H_;
    ulat[i] = user_emb[(size_t)uid[b] * USER_H_ + j];
}

// ---------------------------------------------------------------- kmeans (one block per batch)
__global__ void kmeans_kernel(const float* __restrict__ X, int* __restrict__ ids) {
    __shared__ float cent[KC_ * C_];
    __shared__ float sums[KC_ * C_];
    __shared__ float cnt[KC_];
    __shared__ float csq[KC_];
    int b = blockIdx.x;
    int t = threadIdx.x;
    const float* xb = X + (size_t)b * T_ * C_;
    for (int idx = t; idx < KC_ * C_; idx += blockDim.x) {
        cent[idx] = xb[(size_t)(idx / C_) * C_ + (idx % C_)];
        sums[idx] = 0.f;
    }
    if (t < KC_) cnt[t] = 0.f;
    __syncthreads();
    if (t < KC_) {
        float s = 0.f;
        for (int j = 0; j < C_; j++) { float c = cent[t * C_ + j]; s += c * c; }
        csq[t] = s;
    }
    __syncthreads();
    const float* xr = xb + (size_t)t * C_;
    int myk = 0;
    for (int it = 0; it <= KM_ITERS; ++it) {
        float dot[KC_];
#pragma unroll
        for (int k = 0; k < KC_; k++) dot[k] = 0.f;
        for (int j = 0; j < C_; j++) {
            float xv = xr[j];
#pragma unroll
            for (int k = 0; k < KC_; k++) dot[k] += xv * cent[k * C_ + j];
        }
        float best = csq[0] - 2.f * dot[0];
        myk = 0;
#pragma unroll
        for (int k = 1; k < KC_; k++) {
            float d = csq[k] - 2.f * dot[k];
            if (d < best) { best = d; myk = k; }
        }
        if (it == KM_ITERS) break;
        atomicAdd(&cnt[myk], 1.f);
        for (int j = 0; j < C_; j++) atomicAdd(&sums[myk * C_ + j], xr[j]);
        __syncthreads();
        for (int idx = t; idx < KC_ * C_; idx += blockDim.x) {
            int k = idx / C_;
            float c = cnt[k];
            if (c > 0.f) cent[idx] = sums[idx] / fmaxf(c, 1.f);
            sums[idx] = 0.f;
        }
        __syncthreads();
        if (t < KC_) {
            float s = 0.f;
            for (int j = 0; j < C_; j++) { float c = cent[t * C_ + j]; s += c * c; }
            csq[t] = s;
            cnt[t] = 0.f;
        }
        __syncthreads();
    }
    ids[b * T_ + t] = myk;
}

// ---------------------------------------------------------------- mask + fp32->bf16
__global__ void maskcvt_kernel(float* __restrict__ X, u16* __restrict__ XB,
                               const int* __restrict__ pad) {
    size_t i = (size_t)blockIdx.x * blockDim.x + threadIdx.x;
    if (i >= N_) return;
    int row = (int)(i / C_);
    float v = pad[row] ? 0.f : X[i];
    X[i]  = v;
    XB[i] = f2bf(v);
}

__global__ void cvt_kernel(const float* __restrict__ src, u16* __restrict__ dst, int n) {
    int i = blockIdx.x * blockDim.x + threadIdx.x;
    if (i < n) dst[i] = f2bf(src[i]);
}

// ---------------------------------------------------------------- layernorm (wave32 per row)
__global__ void ln_kernel(const float* __restrict__ X, const float* __restrict__ g,
                          const float* __restrict__ bta, float* __restrict__ outF,
                          u16* __restrict__ outB) {
    int wid  = threadIdx.x >> 5;
    int lane = threadIdx.x & 31;
    int row  = blockIdx.x * 8 + wid;
    const float* xr = X + (size_t)row * C_;
    float v[8];
    float s = 0.f;
#pragma unroll
    for (int j = 0; j < 8; j++) { v[j] = xr[j * 32 + lane]; s += v[j]; }
#pragma unroll
    for (int o = 16; o > 0; o >>= 1) s += __shfl_xor(s, o, 32);
    float mu = s * (1.f / C_);
    float vs = 0.f;
#pragma unroll
    for (int j = 0; j < 8; j++) { float d = v[j] - mu; vs += d * d; }
#pragma unroll
    for (int o = 16; o > 0; o >>= 1) vs += __shfl_xor(vs, o, 32);
    float rstd = rsqrtf(vs * (1.f / C_) + 1e-8f);
#pragma unroll
    for (int j = 0; j < 8; j++) {
        int idx = j * 32 + lane;
        float o = g[idx] * (v[j] - mu) * rstd + bta[idx];
        if (outF) outF[(size_t)row * C_ + idx] = o;
        if (outB) outB[(size_t)row * C_ + idx] = f2bf(o);
    }
}

// ---------------------------------------------------------------- WMMA bf16 GEMM
__global__ void gemm_kernel(const u16* __restrict__ A, const u16* __restrict__ W,
                            const float* __restrict__ bias, const float* __restrict__ res,
                            float* __restrict__ outF, u16* __restrict__ outB,
                            int M, int Nn, int K, int relu) {
    __shared__ __align__(16) u16 Wt[64 * 264];   // W transposed: [n][k]
    __shared__ __align__(16) u16 As[128 * 40];   // A tile: [m][k(32)]
    int tid = threadIdx.x;
    int rb = blockIdx.x * 128;
    int cb = blockIdx.y * 64;

    for (int idx = tid; idx < 64 * K; idx += 256) {
        int n = idx & 63;
        int k = idx >> 6;
        Wt[n * 264 + k] = W[(size_t)k * Nn + cb + n];
    }

    v8f acc[2][2];
    v8f zero = {0.f, 0.f, 0.f, 0.f, 0.f, 0.f, 0.f, 0.f};
#pragma unroll
    for (int i = 0; i < 2; i++)
#pragma unroll
        for (int j = 0; j < 2; j++) acc[i][j] = zero;

    int wid = tid >> 5, lane = tid & 31;
    int wm = wid & 3, wn = wid >> 2;
    int arow0 = wm * 32 + (lane & 15);
    int khalf = (lane >> 4) * 16;
    int bn0   = wn * 32 + (lane & 15);
    int kqb   = (lane >> 4) * 16;
    const u32* WtU = (const u32*)Wt;
    const u32* AsU = (const u32*)As;
    u32* AsW = (u32*)As;
    int lrow = tid >> 1;
    int lcol = (tid & 1) * 16;
    __syncthreads();

    for (int ko = 0; ko < K; ko += 32) {
        const u32* gp = (const u32*)(A + (size_t)(rb + lrow) * K + ko + lcol);
        u32* sp = AsW + (lrow * 40 + lcol) / 2;
#pragma unroll
        for (int j = 0; j < 8; j++) sp[j] = gp[j];
        if (ko + 32 < K)  // warm caches for next tile (global_prefetch_b8)
            __builtin_prefetch(A + (size_t)(rb + lrow) * K + ko + 32 + lcol, 0, 1);
        __syncthreads();

        union { v16bf v; u32 u[8]; } afr[2], bfr[2];
#pragma unroll
        for (int i = 0; i < 2; i++) {
            int row  = arow0 + i * 16;
            int base = row * 20 + (khalf >> 1);
#pragma unroll
            for (int j = 0; j < 4; j++) {
                afr[i].u[j]     = AsU[base + j];
                afr[i].u[4 + j] = AsU[base + 4 + j];
            }
        }
#pragma unroll
        for (int j2 = 0; j2 < 2; j2++) {
            int n    = bn0 + j2 * 16;
            int base = n * 132 + ((ko + kqb) >> 1);
#pragma unroll
            for (int j = 0; j < 8; j++) bfr[j2].u[j] = WtU[base + j];
        }
#pragma unroll
        for (int i = 0; i < 2; i++)
#pragma unroll
            for (int j = 0; j < 2; j++)
                acc[i][j] = __builtin_amdgcn_wmma_f32_16x16x32_bf16(
                    false, afr[i].v, false, bfr[j].v, (short)0, acc[i][j], false, false);
        __syncthreads();
    }

#pragma unroll
    for (int i = 0; i < 2; i++) {
#pragma unroll
        for (int j = 0; j < 2; j++) {
            int n     = cb + wn * 32 + j * 16 + (lane & 15);
            int mbase = rb + wm * 32 + i * 16 + (lane >> 4) * 8;
            float bv = bias ? bias[n] : 0.f;
#pragma unroll
            for (int r = 0; r < 8; r++) {
                size_t idx = (size_t)(mbase + r) * Nn + n;
                float val = acc[i][j][r] + bv;
                if (res)  val += res[idx];
                if (relu) val = fmaxf(val, 0.f);
                if (outF) outF[idx] = val;
                if (outB) outB[idx] = f2bf(val);
            }
        }
    }
}

// ---------------------------------------------------------------- WMMA flash attention
// one block per (b,h); 8 waves; each wave owns 32 query rows (2 q-tiles of 16).
// S = Q K^T via wmma (K-dim = DH = 64), online softmax in C/D layout,
// P bounced through per-wave LDS to A-fragment layout, O += P V via wmma.
__global__ void attn_kernel(const u16* __restrict__ Qb, const u16* __restrict__ Kb,
                            const u16* __restrict__ Vb, const int* __restrict__ ids,
                            const int* __restrict__ pad, u16* __restrict__ Ob) {
    __shared__ __align__(16) u16   Ks[T_ * 72];     // K panel [t][dh], padded
    __shared__ __align__(16) u16   Vt[DH_ * 264];   // V transposed [dh][t], padded
    __shared__ __align__(16) float Ps[8 * 16 * 40]; // per-wave P scratch [q][kk]
    __shared__ int ids_s[T_];
    __shared__ int pad_s[T_];

    int bh = blockIdx.x;
    int b = bh / H_;
    int h = bh % H_;
    int tid = threadIdx.x;
    size_t base = (size_t)b * T_ * C_ + h * DH_;

    for (int idx = tid; idx < T_ * DH_; idx += 256) {
        int t = idx >> 6;
        int d = idx & 63;
        u16 kv = Kb[base + (size_t)t * C_ + d];
        u16 vv = Vb[base + (size_t)t * C_ + d];
        Ks[t * 72 + d]  = kv;
        Vt[d * 264 + t] = vv;
    }
    for (int idx = tid; idx < T_; idx += 256) {
        ids_s[idx] = ids[b * T_ + idx];
        pad_s[idx] = pad[b * T_ + idx];
    }
    __syncthreads();

    int wid = tid >> 5, lane = tid & 31;
    int ln15 = lane & 15;
    int hi   = lane >> 4;           // row-group select in C/D layout
    int khalf = hi * 16;            // A-fragment k start (16-bit layout)
    int kqb   = hi * 16;            // B-fragment k start
    const u32* KsU = (const u32*)Ks;
    const u32* VtU = (const u32*)Vt;
    float* psw = Ps + wid * 16 * 40;
    const float scale = 0.125f;     // 1/sqrt(DH)
    const v8f zero = {0.f, 0.f, 0.f, 0.f, 0.f, 0.f, 0.f, 0.f};

#pragma unroll 1
    for (int qt = wid * 2; qt < wid * 2 + 2; qt++) {
        int q0 = qt * 16;
        // Q A-fragments straight from global (row q per lane, contiguous dh)
        union { v16bf v; u32 u[8]; } aq[2];
#pragma unroll
        for (int ko = 0; ko < 2; ko++) {
            const u32* qp = (const u32*)(Qb + base + (size_t)(q0 + ln15) * C_ + ko * 32);
#pragma unroll
            for (int j = 0; j < 4; j++) {
                aq[ko].u[j]     = qp[(khalf >> 1) + j];
                aq[ko].u[4 + j] = qp[4 + (khalf >> 1) + j];
            }
        }
        int   idq[8];
        float rowOK[8];
#pragma unroll
        for (int r = 0; r < 8; r++) {
            int qrow = q0 + r + hi * 8;
            idq[r]   = ids_s[qrow];
            rowOK[r] = pad_s[qrow] ? 0.f : 1.f;
        }
        float m_run[8], l_run[8];
#pragma unroll
        for (int r = 0; r < 8; r++) { m_run[r] = -1e30f; l_run[r] = 0.f; }
        v8f oacc[4];
#pragma unroll
        for (int d = 0; d < 4; d++) oacc[d] = zero;

#pragma unroll 1
        for (int kb = 0; kb < T_; kb += 32) {
            // scores for 32 keys: two 16x16 tiles, K-dim = 64 -> 2 wmma each
            v8f s0 = zero, s1 = zero;
#pragma unroll
            for (int ko = 0; ko < 2; ko++) {
                union { v16bf v; u32 u[8]; } bk0, bk1;
                int n0 = kb + ln15;
                int n1 = kb + 16 + ln15;
                int koff = (ko * 32 + kqb) >> 1;
#pragma unroll
                for (int j = 0; j < 8; j++) {
                    bk0.u[j] = KsU[n0 * 36 + koff + j];
                    bk1.u[j] = KsU[n1 * 36 + koff + j];
                }
                s0 = __builtin_amdgcn_wmma_f32_16x16x32_bf16(false, aq[ko].v, false, bk0.v, (short)0, s0, false, false);
                s1 = __builtin_amdgcn_wmma_f32_16x16x32_bf16(false, aq[ko].v, false, bk1.v, (short)0, s1, false, false);
            }
            // online softmax (rows live across the 16-lane group)
            int idk0 = ids_s[kb + ln15];
            int idk1 = ids_s[kb + 16 + ln15];
#pragma unroll
            for (int r = 0; r < 8; r++) {
                float ok = rowOK[r];
                float v0 = (ok != 0.f && idk0 == idq[r]) ? s0[r] * scale : -1e9f;
                float v1 = (ok != 0.f && idk1 == idq[r]) ? s1[r] * scale : -1e9f;
                float cmax = rmax16(fmaxf(v0, v1));
                float mnew = fmaxf(m_run[r], cmax);
                float sc = __expf(m_run[r] - mnew);
                float p0 = __expf(v0 - mnew);
                float p1 = __expf(v1 - mnew);
                float rs = rsum16(p0 + p1);
                l_run[r] = l_run[r] * sc + rs;
                m_run[r] = mnew;
#pragma unroll
                for (int d = 0; d < 4; d++) oacc[d][r] *= sc;
                int prow = r + hi * 8;
                psw[prow * 40 + ln15]      = p0;
                psw[prow * 40 + 16 + ln15] = p1;
            }
            // P tile -> A-fragment (bf16), per-wave LDS bounce
            union { v16bf v; u32 u[8]; } ap;
            {
                const float* pr = psw + ln15 * 40;
#pragma unroll
                for (int j = 0; j < 4; j++) {
                    ap.u[j]     = pack2(pr[khalf + 2 * j],      pr[khalf + 2 * j + 1]);
                    ap.u[4 + j] = pack2(pr[8 + khalf + 2 * j], pr[8 + khalf + 2 * j + 1]);
                }
            }
            // O += P @ V (4 dh tiles of 16)
#pragma unroll
            for (int d = 0; d < 4; d++) {
                union { v16bf v; u32 u[8]; } bv;
                int n = d * 16 + ln15;
#pragma unroll
                for (int j = 0; j < 8; j++) bv.u[j] = VtU[n * 132 + ((kb + kqb) >> 1) + j];
                oacc[d] = __builtin_amdgcn_wmma_f32_16x16x32_bf16(false, ap.v, false, bv.v, (short)0, oacc[d], false, false);
            }
        }
        // epilogue: divide by row sums, store bf16
#pragma unroll
        for (int d = 0; d < 4; d++) {
            int dh = d * 16 + ln15;
#pragma unroll
            for (int r = 0; r < 8; r++) {
                int qrow = q0 + r + hi * 8;
                float val = oacc[d][r] / l_run[r];
                Ob[base + (size_t)qrow * C_ + dh] = f2bf(val);
            }
        }
    }
}

// ---------------------------------------------------------------- final logits
__global__ void logits_kernel(const float* __restrict__ OUT, const int* __restrict__ pos,
                              const int* __restrict__ neg, const float* __restrict__ item_emb,
                              const float* __restrict__ ulat, float* __restrict__ dout) {
    int row = blockIdx.x * blockDim.x + threadIdx.x;
    if (row >= BT_) return;
    int b = row / T_;
    const float* o  = OUT + (size_t)row * C_;
    const float* pe = item_emb + (size_t)pos[row] * ITEM_H_;
    const float* ne = item_emb + (size_t)neg[row] * ITEM_H_;
    const float* ul = ulat + (size_t)b * USER_H_;
    float sp = 0.f, sn = 0.f;
    for (int j = 0; j < ITEM_H_; j++) { float ov = o[j]; sp += ov * pe[j]; sn += ov * ne[j]; }
    for (int j = 0; j < USER_H_; j++) { float ov = o[ITEM_H_ + j]; float u = ul[j]; sp += ov * u; sn += ov * u; }
    dout[row] = sp;
    dout[BT_ + row] = sn;
}

// ================================================================ host
extern "C" void kernel_launch(void* const* d_in, const int* in_sizes, int n_in,
                              void* d_out, int out_size, void* d_ws, size_t ws_size,
                              hipStream_t stream) {
    (void)in_sizes; (void)n_in; (void)out_size; (void)ws_size;
    const int*   user_ids = (const int*)d_in[0];
    const int*   seq      = (const int*)d_in[1];
    const int*   pos_seqs = (const int*)d_in[2];
    const int*   neg_seqs = (const int*)d_in[3];
    const float* item_emb = (const float*)d_in[4];
    const float* user_emb = (const float*)d_in[5];
    const float* pos_emb  = (const float*)d_in[6];
    const float* Wq = (const float*)d_in[7];
    const float* Wk = (const float*)d_in[8];
    const float* Wv = (const float*)d_in[9];
    const float* Wo = (const float*)d_in[10];
    const float* ln1_g = (const float*)d_in[11];
    const float* ln1_b = (const float*)d_in[12];
    const float* ln2_g = (const float*)d_in[13];
    const float* ln2_b = (const float*)d_in[14];
    const float* W1 = (const float*)d_in[15];
    const float* b1 = (const float*)d_in[16];
    const float* W2 = (const float*)d_in[17];
    const float* b2 = (const float*)d_in[18];
    const float* lnf_g = (const float*)d_in[19];
    const float* lnf_b = (const float*)d_in[20];
    float* dout = (float*)d_out;

    char* ws = (char*)d_ws;
    size_t off = 0;
    auto take = [&](size_t bytes) -> char* {
        char* p = ws + off;
        off += (bytes + 255) & ~(size_t)255;
        return p;
    };
    float* X    = (float*)take(N_ * 4);
    float* QIN  = (float*)take(N_ * 4);
    u16*   XB   = (u16*)take(N_ * 2);
    u16*   Qb   = (u16*)take(N_ * 2);
    u16*   Kb   = (u16*)take(N_ * 2);
    u16*   Vb   = (u16*)take(N_ * 2);
    u16*   SB0  = (u16*)take(N_ * 2);
    u16*   SB1  = (u16*)take(N_ * 2);
    int*   ids  = (int*)take((size_t)BT_ * 4);
    int*   pad  = (int*)take((size_t)BT_ * 4);
    float* ulat = (float*)take((size_t)B_ * USER_H_ * 4);
    const int WElems = L_ * C_ * C_;
    u16* Wqb = (u16*)take((size_t)WElems * 2);
    u16* Wkb = (u16*)take((size_t)WElems * 2);
    u16* Wvb = (u16*)take((size_t)WElems * 2);
    u16* Wob = (u16*)take((size_t)WElems * 2);
    u16* W1b = (u16*)take((size_t)WElems * 2);
    u16* W2b = (u16*)take((size_t)WElems * 2);

    const int TPB = 256;
    int wblk = (WElems + TPB - 1) / TPB;
    cvt_kernel<<<wblk, TPB, 0, stream>>>(Wq, Wqb, WElems);
    cvt_kernel<<<wblk, TPB, 0, stream>>>(Wk, Wkb, WElems);
    cvt_kernel<<<wblk, TPB, 0, stream>>>(Wv, Wvb, WElems);
    cvt_kernel<<<wblk, TPB, 0, stream>>>(Wo, Wob, WElems);
    cvt_kernel<<<wblk, TPB, 0, stream>>>(W1, W1b, WElems);
    cvt_kernel<<<wblk, TPB, 0, stream>>>(W2, W2b, WElems);

    int nblk = (int)(N_ / TPB);
    embed_kernel<<<nblk, TPB, 0, stream>>>(user_ids, seq, item_emb, user_emb, pos_emb, X, pad);
    ulat_kernel<<<(B_ * USER_H_) / TPB, TPB, 0, stream>>>(user_ids, user_emb, ulat);
    kmeans_kernel<<<B_, TPB, 0, stream>>>(X, ids);
    maskcvt_kernel<<<nblk, TPB, 0, stream>>>(X, XB, pad);

    dim3 ggrid(BT_ / 128, C_ / 64);
    int lnblk = BT_ / 8;
    for (int l = 0; l < L_; l++) {
        const size_t wOff = (size_t)l * C_ * C_;
        ln_kernel<<<lnblk, TPB, 0, stream>>>(X, ln1_g + l * C_, ln1_b + l * C_, QIN, SB0);
        gemm_kernel<<<ggrid, TPB, 0, stream>>>(SB0, Wqb + wOff, nullptr, nullptr, nullptr, Qb, BT_, C_, C_, 0);
        gemm_kernel<<<ggrid, TPB, 0, stream>>>(XB,  Wkb + wOff, nullptr, nullptr, nullptr, Kb, BT_, C_, C_, 0);
        gemm_kernel<<<ggrid, TPB, 0, stream>>>(XB,  Wvb + wOff, nullptr, nullptr, nullptr, Vb, BT_, C_, C_, 0);
        attn_kernel<<<B_ * H_, TPB, 0, stream>>>(Qb, Kb, Vb, ids, pad, SB0);
        gemm_kernel<<<ggrid, TPB, 0, stream>>>(SB0, Wob + wOff, nullptr, QIN, X, nullptr, BT_, C_, C_, 0);
        maskcvt_kernel<<<nblk, TPB, 0, stream>>>(X, XB, pad);
        ln_kernel<<<lnblk, TPB, 0, stream>>>(X, ln2_g + l * C_, ln2_b + l * C_, nullptr, SB0);
        gemm_kernel<<<ggrid, TPB, 0, stream>>>(SB0, W1b + wOff, b1 + l * C_, nullptr, nullptr, SB1, BT_, C_, C_, 1);
        gemm_kernel<<<ggrid, TPB, 0, stream>>>(SB1, W2b + wOff, b2 + l * C_, X, X, nullptr, BT_, C_, C_, 0);
        maskcvt_kernel<<<nblk, TPB, 0, stream>>>(X, XB, pad);
    }
    ln_kernel<<<lnblk, TPB, 0, stream>>>(X, lnf_g, lnf_b, QIN, nullptr);
    logits_kernel<<<BT_ / TPB, TPB, 0, stream>>>(QIN, pos_seqs, neg_seqs, item_emb, ulat, dout);
}
